// XformersModule_28905129902834
// MI455X (gfx1250) — matrix operand order repaired and checked by reading, not compiled, not run on hardware
//
#include <hip/hip_runtime.h>

// ---------------------------------------------------------------------------
// CDNA5 (gfx1250, wave32) implementation of the token-attention module.
// All GEMMs use V_WMMA_F32_16X16X32_BF16 (bf16 in, f32 accumulate) with
// 32x64 register blocking per wave (8 WMMAs / 6 fragment loads per K-step), 32x32 in k_final.
// ---------------------------------------------------------------------------

typedef __bf16 bf16;
typedef __attribute__((ext_vector_type(16))) __bf16 v16bf;
typedef __attribute__((ext_vector_type(8)))  float   v8f;

#define DICTSIZE 256
#define NUMCHAR  1024
#define EMB      512
#define BATCH    128
#define ROWS     (BATCH * NUMCHAR)          // 131072
#define KTOT     ((long)NUMCHAR * EMB)      // 524288 (flattened aff row length)

// ---------------------------------------------------------------------------
// WMMA fragment helpers (layouts per CDNA5 ISA 7.12.2, wave32)
// ---------------------------------------------------------------------------

// A fragment: 16x32 bf16 tile, row-major source with row stride `ld` (elems).
// Per lane this is one contiguous 32-byte run (k = 16*(lane/16) + s) -> 2x b128.
static __device__ __forceinline__ v16bf load_A_frag(const bf16* base, long ld, int lane) {
  const bf16* p = base + (long)(lane & 15) * ld;
  const int koff = (lane & 16) ? 16 : 0;
  v16bf a;
#pragma unroll
  for (int v = 0; v < 8; ++v) {
    const int k0 = (v << 1) + koff;
    a[2 * v]     = p[k0];
    a[2 * v + 1] = p[k0 + 1];
  }
  return a;
}

// B fragment: 32x16 (KxN) where memory holds B^T row-major, i.e. source[n][k]
// with row stride `ld` (k contiguous): one 32-byte run per lane -> 2x b128.
static __device__ __forceinline__ v16bf load_B_kcontig(const bf16* base, long ld, int lane) {
  const bf16* p = base + (long)(lane & 15) * ld + ((lane & 16) ? 16 : 0);
  v16bf b;
#pragma unroll
  for (int v = 0; v < 16; ++v) b[v] = p[v];
  return b;
}

// Same, but source is f32 (W_lin streamed once + converted in-register).
static __device__ __forceinline__ v16bf load_B_kcontig_f32(const float* base, long ld, int lane) {
  const float* p = base + (long)(lane & 15) * ld + ((lane & 16) ? 16 : 0);
  v16bf b;
#pragma unroll
  for (int v = 0; v < 16; ++v) b[v] = (bf16)p[v];
  return b;
}

// C/D 16x16 f32 store helpers: VGPR r -> row M=r (lanes 0-15) / M=r+8 (16-31).
static __device__ __forceinline__ void store_C_f32(float* base, long ld, int lane, v8f c) {
  float* p = base + (long)((lane & 16) ? 8 : 0) * ld + (lane & 15);
#pragma unroll
  for (int r = 0; r < 8; ++r) p[(long)r * ld] = c[r];
}

static __device__ __forceinline__ void store_C_bf16(bf16* base, long ld, int lane, v8f c) {
  bf16* p = base + (long)((lane & 16) ? 8 : 0) * ld + (lane & 15);
#pragma unroll
  for (int r = 0; r < 8; ++r) p[(long)r * ld] = (bf16)c[r];
}

static __device__ __forceinline__ v8f wmma_bf16(v16bf a, v16bf b, v8f c) {
  return __builtin_amdgcn_wmma_f32_16x16x32_bf16(false, a, false, b, (short)0, c, false, false);
}

// ---------------------------------------------------------------------------
// Kernel 1: gather token embeddings and convert to bf16.  One row per block.
// ---------------------------------------------------------------------------
__global__ __launch_bounds__(256) void k_gather(const int* __restrict__ tok,
                                                const float* __restrict__ table,
                                                bf16* __restrict__ emb) {
  const long row = blockIdx.x;                  // 0..131071
  const float* src = table + (long)tok[row] * EMB;
  bf16* dst = emb + row * EMB;
  for (int i = threadIdx.x; i < EMB; i += blockDim.x) dst[i] = (bf16)src[i];
}

// Kernel 1b: convert Wq, Wk to bf16 (one elementwise pass each).
__global__ __launch_bounds__(256) void k_cvt_w(const float* __restrict__ wq,
                                               const float* __restrict__ wk,
                                               bf16* __restrict__ wqb,
                                               bf16* __restrict__ wkb) {
  const int i = blockIdx.x * blockDim.x + threadIdx.x;
  if (i < EMB * EMB) { wqb[i] = (bf16)wq[i]; wkb[i] = (bf16)wk[i]; }
}

// ---------------------------------------------------------------------------
// Kernel 1c: per-batch transpose emb[b][m][e] -> embT[b][e][m] via LDS tiles,
// so the aff GEMM's B operand becomes k-contiguous (b128 loads, no scalar
// gathers).  32x32 bf16 tiles, 256 threads, padded LDS to dodge bank conflicts.
// ---------------------------------------------------------------------------
__global__ __launch_bounds__(256) void k_transpose(const bf16* __restrict__ emb,
                                                   bf16* __restrict__ embT) {
  __shared__ bf16 t[32][33];
  const int blk = blockIdx.x;                   // 128 * 32 * 16
  const int et = blk & 15;
  const int r  = blk >> 4;
  const int mt = r & 31;
  const int bi = r >> 5;
  const int lx = threadIdx.x & 31, ly = threadIdx.x >> 5;   // 32 x 8
  const bf16* src = emb + ((long)bi * NUMCHAR + mt * 32) * EMB + et * 32;
#pragma unroll
  for (int rr = 0; rr < 4; ++rr)
    t[ly + 8 * rr][lx] = src[(long)(ly + 8 * rr) * EMB + lx];
  __syncthreads();
  bf16* dst = embT + ((long)bi * EMB + et * 32) * NUMCHAR + mt * 32;
#pragma unroll
  for (int rr = 0; rr < 4; ++rr)
    dst[(long)(ly + 8 * rr) * NUMCHAR + lx] = t[lx][ly + 8 * rr];
}

// ---------------------------------------------------------------------------
// Kernel 2: projection  out[m,n] = sum_e emb[m,e] * W[n,e]
// M=131072, N=512, K=512.  32(M) x 64(N) per wave: 2 A-frags x 4 B-frags.
// ---------------------------------------------------------------------------
__global__ __launch_bounds__(128) void k_proj(const bf16* __restrict__ emb,
                                              const bf16* __restrict__ W,
                                              bf16* __restrict__ out) {
  const int wave = threadIdx.x >> 5, lane = threadIdx.x & 31;
  const long tile = (long)blockIdx.x * 4 + wave;   // 32768 wave-tiles
  const int  nq = (int)(tile & 7);                 // 8 N-quads of 64
  const long mp = tile >> 3;                       // 4096 M-pairs of 32
  const bf16* abase = emb + mp * 32 * EMB;
  const bf16* bbase = W + (long)nq * 64 * EMB;
  v8f c[2][4] = {};
  for (int kc = 0; kc < EMB; kc += 32) {
    v16bf a0 = load_A_frag(abase + kc, EMB, lane);
    v16bf a1 = load_A_frag(abase + 16 * EMB + kc, EMB, lane);
#pragma unroll
    for (int j = 0; j < 4; ++j) {
      v16bf b = load_B_kcontig(bbase + (long)j * 16 * EMB + kc, EMB, lane);
      c[0][j] = wmma_bf16(a0, b, c[0][j]);
      c[1][j] = wmma_bf16(a1, b, c[1][j]);
    }
  }
#pragma unroll
  for (int i = 0; i < 2; ++i)
#pragma unroll
    for (int j = 0; j < 4; ++j)
      store_C_bf16(out + (mp * 32 + i * 16) * EMB + nq * 64 + j * 16, EMB, lane, c[i][j]);
}

// ---------------------------------------------------------------------------
// Kernel 3: scores[b,qi,mj] = sum_k q[b,qi,k]*k[b,mj,k]
// 32(Q) x 64(Kpos) per wave; coarse causal skip (jq*64 > qp*32+31).
// ---------------------------------------------------------------------------
__global__ __launch_bounds__(128) void k_scores(const bf16* __restrict__ q,
                                                const bf16* __restrict__ kk,
                                                float* __restrict__ S) {
  const int wave = threadIdx.x >> 5, lane = threadIdx.x & 31;
  const long tile = (long)blockIdx.x * 4 + wave;   // bi*512 + qp*16 + jq
  const int  jq = (int)(tile & 15);                // 16 key-quads of 64
  const long r  = tile >> 4;
  const int  qp = (int)(r & 31);                   // 32 query-pairs of 32
  const int  bi = (int)(r >> 5);
  if (jq * 64 > qp * 32 + 31) return;              // fully masked (wave-uniform)
  const bf16* qb = q  + ((long)bi * NUMCHAR + qp * 32) * EMB;
  const bf16* kb = kk + ((long)bi * NUMCHAR + jq * 64) * EMB;
  v8f c[2][4] = {};
  for (int kc = 0; kc < EMB; kc += 32) {
    v16bf a0 = load_A_frag(qb + kc, EMB, lane);
    v16bf a1 = load_A_frag(qb + 16 * EMB + kc, EMB, lane);
#pragma unroll
    for (int j = 0; j < 4; ++j) {
      v16bf b = load_B_kcontig(kb + (long)j * 16 * EMB + kc, EMB, lane);
      c[0][j] = wmma_bf16(a0, b, c[0][j]);
      c[1][j] = wmma_bf16(a1, b, c[1][j]);
    }
  }
  float* sbase = S + ((long)bi * NUMCHAR + qp * 32) * NUMCHAR + (long)jq * 64;
#pragma unroll
  for (int i = 0; i < 2; ++i)
#pragma unroll
    for (int j = 0; j < 4; ++j)
      store_C_f32(sbase + (long)i * 16 * NUMCHAR + j * 16, NUMCHAR, lane, c[i][j]);
}

// ---------------------------------------------------------------------------
// Kernel 4: causal softmax over each score row (length 1024, valid = qi+1).
// Reads f32 row, writes bf16 probabilities IN PLACE (row pitch stays 4096 B),
// zeros for masked positions so downstream GEMM can read rectangular tiles.
// ---------------------------------------------------------------------------
__global__ __launch_bounds__(256) void k_softmax(float* __restrict__ S) {
  __shared__ float red[256];
  const long row = blockIdx.x;                      // 0..131071
  const int  qi  = (int)(row & (NUMCHAR - 1));
  float* srow = S + row * NUMCHAR;
  bf16*  prow = (bf16*)srow;                        // in-place, pitched

  float v[4], mx = -1e30f;
#pragma unroll
  for (int j = 0; j < 4; ++j) {
    const int idx = threadIdx.x + j * 256;
    v[j] = (idx <= qi) ? srow[idx] : -1e30f;
    mx = fmaxf(mx, v[j]);
  }
  red[threadIdx.x] = mx; __syncthreads();
  for (int s = 128; s > 0; s >>= 1) {
    if (threadIdx.x < s) red[threadIdx.x] = fmaxf(red[threadIdx.x], red[threadIdx.x + s]);
    __syncthreads();
  }
  mx = red[0]; __syncthreads();

  float e[4], sum = 0.f;
#pragma unroll
  for (int j = 0; j < 4; ++j) {
    const int idx = threadIdx.x + j * 256;
    e[j] = (idx <= qi) ? __expf(v[j] - mx) : 0.f;
    sum += e[j];
  }
  red[threadIdx.x] = sum; __syncthreads();
  for (int s = 128; s > 0; s >>= 1) {
    if (threadIdx.x < s) red[threadIdx.x] += red[threadIdx.x + s];
    __syncthreads();
  }
  const float inv = 1.0f / red[0];
#pragma unroll
  for (int j = 0; j < 4; ++j) {
    const int idx = threadIdx.x + j * 256;
    prow[idx] = (bf16)(e[j] * inv);                 // all f32 reads done pre-barrier
  }
}

// ---------------------------------------------------------------------------
// Kernel 5: aff[b,qi,e] = sum_m P[b,qi,m] * embT[b,e,m]   (causal-truncated K)
// P is bf16 with row pitch 2*NUMCHAR elements (overlaid on f32 score rows).
// 32(Q) x 64(E) per wave; K loop runs exactly qp+1 chunks of 32 keys.
// ---------------------------------------------------------------------------
__global__ __launch_bounds__(128) void k_aff(const float* __restrict__ Sraw,
                                             const bf16* __restrict__ embT,
                                             bf16* __restrict__ aff) {
  const int wave = threadIdx.x >> 5, lane = threadIdx.x & 31;
  const long tile = (long)blockIdx.x * 4 + wave;    // bi*256 + qp*8 + eq
  const int  eq = (int)(tile & 7);                  // 8 E-quads of 64
  const long r  = tile >> 3;
  const int  qp = (int)(r & 31);                    // 32 query-pairs of 32
  const int  bi = (int)(r >> 5);
  const bf16* P = (const bf16*)Sraw;
  const long PPITCH = 2L * NUMCHAR;                 // bf16 elems per pitched row
  const bf16* arow = P + ((long)bi * NUMCHAR + qp * 32) * PPITCH;
  const bf16* bcol = embT + ((long)bi * EMB + eq * 64) * NUMCHAR;
  const int nch = qp + 1;                           // 32-key chunks with nonzero P
  v8f c[2][4] = {};
  for (int j = 0; j < nch; ++j) {
    v16bf a0 = load_A_frag(arow + j * 32, PPITCH, lane);
    v16bf a1 = load_A_frag(arow + 16 * PPITCH + j * 32, PPITCH, lane);
#pragma unroll
    for (int t = 0; t < 4; ++t) {
      v16bf b = load_B_kcontig(bcol + (long)t * 16 * NUMCHAR + j * 32, NUMCHAR, lane);
      c[0][t] = wmma_bf16(a0, b, c[0][t]);
      c[1][t] = wmma_bf16(a1, b, c[1][t]);
    }
  }
#pragma unroll
  for (int i = 0; i < 2; ++i)
#pragma unroll
    for (int t = 0; t < 4; ++t)
      store_C_bf16(aff + ((long)bi * NUMCHAR + qp * 32 + i * 16) * EMB + eq * 64 + t * 16,
                   EMB, lane, c[i][t]);
}

// ---------------------------------------------------------------------------
// Kernel 6a: out[m,d] = b_lin[d]  (accumulation base for split-K atomics)
// ---------------------------------------------------------------------------
__global__ __launch_bounds__(256) void k_init_out(float* __restrict__ out,
                                                  const float* __restrict__ b_lin) {
  const int i = blockIdx.x * blockDim.x + threadIdx.x;   // 32768
  out[i] = b_lin[i & (DICTSIZE - 1)];
}

// ---------------------------------------------------------------------------
// Kernel 6b: out += aff_flat[128,524288] @ W_lin^T[524288,256]
// Split-K: block = (kchunk of 4096, quarter); 8 waves x 4 blocks cover the
// 4x8 grid of 32x32 tiles.  W_lin stays f32 in HBM (read exactly once),
// converted in-register, with prefetch-ahead on the 512 MB stream.
// ---------------------------------------------------------------------------
__global__ __launch_bounds__(256) void k_final(const bf16* __restrict__ aff,
                                               const float* __restrict__ wlin,
                                               float* __restrict__ out) {
  const int wave = threadIdx.x >> 5, lane = threadIdx.x & 31;
  const int kchunk = blockIdx.x >> 2;               // 128 chunks of 4096
  const int flat   = (blockIdx.x & 3) * 8 + wave;   // 0..31 wave-tiles
  const int mp = flat & 3;                          // 4 M-pairs (32 rows)
  const int np = flat >> 2;                         // 8 N-pairs (32 cols)
  const long k0 = (long)kchunk * 4096;
  const bf16*  abase = aff  + (long)mp * 32 * KTOT;
  const float* bbase = wlin + (long)np * 32 * KTOT;
  v8f c[2][2] = {};
  for (long kc = k0; kc < k0 + 4096; kc += 32) {
    __builtin_prefetch((const void*)(bbase + kc + 1024), 0, 1);   // W_lin stream-ahead
    v16bf a0 = load_A_frag(abase + kc, KTOT, lane);
    v16bf a1 = load_A_frag(abase + 16 * KTOT + kc, KTOT, lane);
    v16bf b0 = load_B_kcontig_f32(bbase + kc, KTOT, lane);
    v16bf b1 = load_B_kcontig_f32(bbase + 16 * KTOT + kc, KTOT, lane);
    c[0][0] = wmma_bf16(a0, b0, c[0][0]);
    c[0][1] = wmma_bf16(a0, b1, c[0][1]);
    c[1][0] = wmma_bf16(a1, b0, c[1][0]);
    c[1][1] = wmma_bf16(a1, b1, c[1][1]);
  }
  const int n = lane & 15, mb = (lane & 16) ? 8 : 0;
#pragma unroll
  for (int i = 0; i < 2; ++i)
#pragma unroll
    for (int j = 0; j < 2; ++j)
#pragma unroll
      for (int r = 0; r < 8; ++r)
        atomicAdd(&out[(mp * 32 + i * 16 + mb + r) * DICTSIZE + np * 32 + j * 16 + n],
                  c[i][j][r]);
}

// ---------------------------------------------------------------------------
// Host-side launcher
// ---------------------------------------------------------------------------
extern "C" void kernel_launch(void* const* d_in, const int* in_sizes, int n_in,
                              void* d_out, int out_size, void* d_ws, size_t ws_size,
                              hipStream_t stream) {
  (void)in_sizes; (void)n_in; (void)out_size; (void)ws_size;
  const int*   tok    = (const int*)d_in[0];
  const float* table  = (const float*)d_in[1];
  const float* Wq     = (const float*)d_in[2];
  const float* Wk     = (const float*)d_in[3];
  // d_in[4] (Wv) is computed-but-unused in the reference: skipped entirely.
  const float* W_lin  = (const float*)d_in[5];
  const float* b_lin  = (const float*)d_in[6];
  float* out = (float*)d_out;

  // Workspace layout (bytes):
  char* ws = (char*)d_ws;
  const long MB128 = 134217728L;
  bf16*  emb  = (bf16*)(ws);                               // 128 MiB
  bf16*  q    = (bf16*)(ws + 1 * MB128);                   // 128 MiB
  bf16*  k    = (bf16*)(ws + 2 * MB128);                   // 128 MiB
  bf16*  embT = (bf16*)(ws + 3 * MB128);                   // 128 MiB
  bf16*  Wqb  = (bf16*)(ws + 4 * MB128);                   // 0.5 MiB
  bf16*  Wkb  = (bf16*)(ws + 4 * MB128 + 524288L);         // 0.5 MiB
  float* S    = (float*)(ws + 4 * MB128 + 1048576L);       // 512 MiB (P overlaid)
  bf16*  aff  = (bf16*)(ws + 4 * MB128 + 1048576L + 536870912L); // 128 MiB

  // 1) embed gather + bf16 convert; weight convert; emb transpose
  k_gather<<<ROWS, 256, 0, stream>>>(tok, table, emb);
  k_cvt_w<<<(EMB * EMB + 255) / 256, 256, 0, stream>>>(Wq, Wk, Wqb, Wkb);
  k_transpose<<<BATCH * 32 * 16, 256, 0, stream>>>(emb, embT);

  // 2) q/k projections: 32768 wave-tiles each, 4 waves/block
  k_proj<<<8192, 128, 0, stream>>>(emb, Wqb, q);
  k_proj<<<8192, 128, 0, stream>>>(emb, Wkb, k);

  // 3) causal scores (f32)
  k_scores<<<16384, 128, 0, stream>>>(q, k, S);

  // 4) causal softmax -> bf16 P in place
  k_softmax<<<ROWS, 256, 0, stream>>>(S);

  // 5) aff = P @ emb (bf16, via embT)
  k_aff<<<8192, 128, 0, stream>>>(S, embT, aff);

  // 6) out = aff_flat @ W_lin^T + b_lin  (split-K, f32 atomics)
  k_init_out<<<(BATCH * DICTSIZE) / 256, 256, 0, stream>>>(out, b_lin);
  k_final<<<512, 256, 0, stream>>>(aff, W_lin, out);
}
